// MultiHeadAttention_4518305595721
// MI455X (gfx1250) — matrix operand order repaired and checked by reading, not compiled, not run on hardware
//
#include <hip/hip_runtime.h>
#include <hip/hip_bf16.h>

// ---------------------------------------------------------------------------
// MHA forward for MI455X (gfx1250, wave32, WMMA).
// cast f32->f16 -> 3x WMMA GEMM (QKV; 128x128 block tile, double-buffered
// async-to-LDS staging overlapped with WMMA) -> RoPE -> causal flash
// attention (WMMA QK^T / PV, online softmax, ds_swizzle reductions,
// double-buffered async KV stream) -> WMMA GEMM out proj (f32 out).
// ---------------------------------------------------------------------------

typedef __attribute__((ext_vector_type(16))) _Float16 half16;
typedef __attribute__((ext_vector_type(8)))  float    float8;

#define WMMA_F16F32(a, b, c) \
  __builtin_amdgcn_wmma_f32_16x16x32_f16(false, (a), false, (b), (short)0, (c), false, false)

static constexpr int   kB      = 2;
static constexpr int   kS      = 2048;
static constexpr int   kD      = 2048;
static constexpr int   kH      = 16;
static constexpr int   kHD     = 128;
static constexpr float kScale  = 0.08838834764831845f;   // 1/sqrt(128)
static constexpr float kNegBig = -1.0e30f;

// --- CDNA5 async global->LDS (ASYNCcnt path), per-lane 16B ------------------
__device__ inline uint32_t lds_off_of(const void* p) {
  // generic shared pointer: low 32 bits are the LDS byte offset
  return (uint32_t)(uintptr_t)p;
}
__device__ inline void async_load_b128(uint32_t lds_off, const void* gptr) {
  unsigned long long ga = (unsigned long long)(uintptr_t)gptr;
  asm volatile("global_load_async_to_lds_b128 %0, %1, off"
               :: "v"(lds_off), "v"(ga) : "memory");
}
__device__ inline void wait_async0() {
  asm volatile("s_wait_asynccnt 0" ::: "memory");
}
__device__ inline void wait_ds0() {
  asm volatile("s_wait_dscnt 0" ::: "memory");
}

// --- 16-lane xor-swizzle reductions (constant-pattern ds_swizzle) -----------
__device__ inline float reduce_max16(float x) {
  x = fmaxf(x, __int_as_float(__builtin_amdgcn_ds_swizzle(__float_as_int(x), 0x041f)));
  x = fmaxf(x, __int_as_float(__builtin_amdgcn_ds_swizzle(__float_as_int(x), 0x081f)));
  x = fmaxf(x, __int_as_float(__builtin_amdgcn_ds_swizzle(__float_as_int(x), 0x101f)));
  x = fmaxf(x, __int_as_float(__builtin_amdgcn_ds_swizzle(__float_as_int(x), 0x201f)));
  return x;
}
__device__ inline float reduce_sum16(float x) {
  x += __int_as_float(__builtin_amdgcn_ds_swizzle(__float_as_int(x), 0x041f));
  x += __int_as_float(__builtin_amdgcn_ds_swizzle(__float_as_int(x), 0x081f));
  x += __int_as_float(__builtin_amdgcn_ds_swizzle(__float_as_int(x), 0x101f));
  x += __int_as_float(__builtin_amdgcn_ds_swizzle(__float_as_int(x), 0x201f));
  return x;
}

// ---------------------------------------------------------------------------
// f32 -> f16 cast (grid-stride)
// ---------------------------------------------------------------------------
__global__ void cast_f32_to_f16(const float* __restrict__ src,
                                _Float16* __restrict__ dst, int n) {
  int i = blockIdx.x * blockDim.x + threadIdx.x;
  int stride = gridDim.x * blockDim.x;
  for (; i < n; i += stride) dst[i] = (_Float16)src[i];
}

// ---------------------------------------------------------------------------
// RoPE applied in place to a [B,S,H,hd] f16 tensor (interleaved pairs).
// ---------------------------------------------------------------------------
__global__ void rope_kernel(_Float16* __restrict__ t, int S, int total) {
  int u = blockIdx.x * blockDim.x + threadIdx.x;
  int stride = gridDim.x * blockDim.x;
  const float ln_base_over_hd = 9.210340371976184f / 128.0f; // ln(10000)/hd
  for (; u < total; u += stride) {
    int i   = u & 63;          // pair index within head
    int h   = (u >> 6) & 15;   // head
    int row = u >> 10;         // b*S + s
    int s   = row % S;
    size_t base = (size_t)row * kD + (size_t)h * kHD + 2 * i;
    float x1 = (float)t[base];
    float x2 = (float)t[base + 1];
    float inv_freq = __expf(-(float)(2 * i) * ln_base_over_hd);
    float ang = (float)s * inv_freq;
    float sn, cs;
    __sincosf(ang, &sn, &cs);
    t[base]     = (_Float16)(x1 * cs - x2 * sn);
    t[base + 1] = (_Float16)(x1 * sn + x2 * cs);
  }
}

// ---------------------------------------------------------------------------
// Fragment gathers from LDS tiles (CDNA5 16-bit WMMA VGPR layouts).
// Both gathers are 2x16B contiguous per lane -> compiler emits ds_load_b128.
// ---------------------------------------------------------------------------
__device__ inline half16 load_a_frag(const _Float16* base, int ld, int lane) {
  half16 a;
  int row = lane & 15, kh = lane >> 4;
#pragma unroll
  for (int j = 0; j < 8; ++j) {
    int kk = ((j >> 2) << 4) + kh * 8 + ((j & 3) << 1);
    a[2 * j]     = base[row * ld + kk];
    a[2 * j + 1] = base[row * ld + kk + 1];
  }
  return a;
}

__device__ inline half16 load_b_frag(const _Float16* base, int ld, int lane) {
  half16 bf;
  int col = lane & 15, kh = lane >> 4;
#pragma unroll
  for (int j = 0; j < 8; ++j) {
    int kk = ((j >> 2) << 4) + kh * 8 + ((j & 3) << 1);
    bf[2 * j]     = base[col * ld + kk];
    bf[2 * j + 1] = base[col * ld + kk + 1];
  }
  return bf;
}

// ---------------------------------------------------------------------------
// C[M,N] = A[M,K] * B[N,K]^T   ('bsd,ed->bse': B rows = out features).
// 128 threads = 4 waves; block tile 128x128; wave tile 64x64 (16 wmma per
// K-step). K staged 32-wide through a 2-deep LDS pipeline: wait chunk i,
// barrier, issue async for chunk i+1 (other buffer), compute chunk i while
// the async engine fills i+1. One barrier per K-step.
// ---------------------------------------------------------------------------
template <typename OutT>
__global__ __launch_bounds__(128) void gemm_nt_wmma(
    const _Float16* __restrict__ A, const _Float16* __restrict__ Bm,
    OutT* __restrict__ C, int M, int N, int K) {
  __shared__ _Float16 sA[2][128][40];
  __shared__ _Float16 sB[2][128][40];

  const int tid  = threadIdx.x;
  const int lane = tid & 31;
  const int w    = tid >> 5;
  const int m0 = blockIdx.y * 128, n0 = blockIdx.x * 128;
  const int wm = (w >> 1) * 64, wn = (w & 1) * 64;

  float8 acc[4][4] = {};

  // Cooperative staging: thread `tid` owns row `tid` of both 128x32 tiles.
  const _Float16* gA = &A[(size_t)(m0 + tid) * K];
  const _Float16* gB = &Bm[(size_t)(n0 + tid) * K];
  const uint32_t laA[2] = { lds_off_of(&sA[0][tid][0]), lds_off_of(&sA[1][tid][0]) };
  const uint32_t laB[2] = { lds_off_of(&sB[0][tid][0]), lds_off_of(&sB[1][tid][0]) };

  auto stage = [&](int k0, int buf) {
#pragma unroll
    for (int t = 0; t < 4; ++t) {
      async_load_b128(laA[buf] + t * 16, gA + k0 + t * 8);
      async_load_b128(laB[buf] + t * 16, gB + k0 + t * 8);
    }
  };

  const int nSteps = K / 32;
  stage(0, 0);

  for (int s = 0; s < nSteps; ++s) {
    const int buf = s & 1;
    wait_async0();        // chunk s resident in LDS buf
    __syncthreads();      // all waves done reading buf^1 (iter s-1) + see data
    if (s + 1 < nSteps) {
      stage((s + 1) * 32, buf ^ 1);           // overlap DMA with WMMA below
      if (s + 2 < nSteps) {
        __builtin_prefetch(gA + (s + 2) * 32, 0, 3);
        __builtin_prefetch(gB + (s + 2) * 32, 0, 3);
      }
    }

    half16 af[4], bf[4];
#pragma unroll
    for (int mt = 0; mt < 4; ++mt) af[mt] = load_a_frag(&sA[buf][wm + mt * 16][0], 40, lane);
#pragma unroll
    for (int nt = 0; nt < 4; ++nt) bf[nt] = load_b_frag(&sB[buf][wn + nt * 16][0], 40, lane);

#pragma unroll
    for (int mt = 0; mt < 4; ++mt)
#pragma unroll
      for (int nt = 0; nt < 4; ++nt)
        acc[mt][nt] = WMMA_F16F32(af[mt], bf[nt], acc[mt][nt]);
  }

  // C/D layout: lanes 0-15 hold N=lane, M=i; lanes 16-31 hold N=lane-16, M=8+i
  const int cn = lane & 15, hh = lane >> 4;
#pragma unroll
  for (int mt = 0; mt < 4; ++mt)
#pragma unroll
    for (int nt = 0; nt < 4; ++nt)
#pragma unroll
      for (int i = 0; i < 8; ++i) {
        int m = m0 + wm + mt * 16 + hh * 8 + i;
        int n = n0 + wn + nt * 16 + cn;
        C[(size_t)m * N + n] = (OutT)acc[mt][nt][i];
      }
}

// ---------------------------------------------------------------------------
// Flash attention, causal. Grid (S/64, H, B), 128 threads = 4 waves.
// Each wave owns 16 query rows; KV streamed in 32-row chunks through a
// 2-deep async LDS pipeline. Scores: 2 WMMA accumulators chained over
// hd=128. Online softmax with ds_swizzle xor reductions in 16-lane halves.
// P bounced through wave-private LDS into the A-fragment layout; PV = 8
// WMMA tiles covering the 128 output columns.
// ---------------------------------------------------------------------------
__global__ __launch_bounds__(128) void attn_fwd(
    const _Float16* __restrict__ Q, const _Float16* __restrict__ Kt,
    const _Float16* __restrict__ Vt, _Float16* __restrict__ ctx, int S) {
  __shared__ _Float16 Ks[2][32][136];
  __shared__ _Float16 Vs[2][32][136];
  __shared__ _Float16 Ps[4][16][32];

  const int tid  = threadIdx.x;
  const int lane = tid & 31;
  const int w    = tid >> 5;
  const int bb   = blockIdx.z;
  const int h    = blockIdx.y;
  const int qblk = blockIdx.x * 64;
  const int q0   = qblk + w * 16;
  const size_t headOff = (size_t)h * kHD;

  const int cn = lane & 15;   // column-within-tile / A row
  const int kh = lane >> 4;

  // Load Q fragments for this wave's 16 rows (once).
  half16 qf[4];
  {
    const _Float16* qb = Q + ((size_t)bb * S + q0) * kD + headOff;
#pragma unroll
    for (int kg = 0; kg < 4; ++kg)
#pragma unroll
      for (int j = 0; j < 8; ++j) {
        int kk = kg * 32 + ((j >> 2) << 4) + kh * 8 + ((j & 3) << 1);
        qf[kg][2 * j]     = qb[(size_t)cn * kD + kk];
        qf[kg][2 * j + 1] = qb[(size_t)cn * kD + kk + 1];
      }
  }

  float8 o[8] = {};
  float mrow[8], lrow[8];
#pragma unroll
  for (int i = 0; i < 8; ++i) { mrow[i] = kNegBig; lrow[i] = 0.0f; }

  // KV staging: thread owns row tid>>2, 32-col segment (tid&3)*32.
  const int r  = tid >> 2;
  const int c0 = (tid & 3) * 32;
  const _Float16* gK = Kt + ((size_t)bb * S + r) * kD + headOff + c0;
  const _Float16* gV = Vt + ((size_t)bb * S + r) * kD + headOff + c0;
  const uint32_t lK[2] = { lds_off_of(&Ks[0][r][c0]), lds_off_of(&Ks[1][r][c0]) };
  const uint32_t lV[2] = { lds_off_of(&Vs[0][r][c0]), lds_off_of(&Vs[1][r][c0]) };

  auto stageKV = [&](int c, int buf) {
    const size_t gOff = (size_t)c * 32 * kD;
#pragma unroll
    for (int t = 0; t < 4; ++t) {
      async_load_b128(lK[buf] + t * 16, gK + gOff + t * 8);
      async_load_b128(lV[buf] + t * 16, gV + gOff + t * 8);
    }
  };

  const int nChunks = (qblk + 64) / 32;   // chunks needed by last row of block
  stageKV(0, 0);

  for (int c = 0; c < nChunks; ++c) {
    const int kv0 = c * 32;
    const int buf = c & 1;
    wait_async0();
    __syncthreads();
    if (c + 1 < nChunks) {
      stageKV(c + 1, buf ^ 1);              // overlap DMA with WMMA below
      if (c + 2 < nChunks) {
        __builtin_prefetch(gK + (size_t)(c + 2) * 32 * kD, 0, 3);
        __builtin_prefetch(gV + (size_t)(c + 2) * 32 * kD, 0, 3);
      }
    }

    if (kv0 <= q0 + 15) {   // wave-uniform: chunk touches this wave's rows
      // ---- scores: S = Q * K^T  (two 16-col groups) ----
      float8 s0 = {};
      float8 s1 = {};
#pragma unroll
      for (int kg = 0; kg < 4; ++kg) {
        half16 bk0, bk1;
#pragma unroll
        for (int j = 0; j < 8; ++j) {
          int kk = kg * 32 + ((j >> 2) << 4) + kh * 8 + ((j & 3) << 1);
          bk0[2 * j]     = Ks[buf][cn][kk];
          bk0[2 * j + 1] = Ks[buf][cn][kk + 1];
          bk1[2 * j]     = Ks[buf][16 + cn][kk];
          bk1[2 * j + 1] = Ks[buf][16 + cn][kk + 1];
        }
        s0 = WMMA_F16F32(qf[kg], bk0, s0);
        s1 = WMMA_F16F32(qf[kg], bk1, s1);
      }

      // ---- online softmax (per accumulator row) ----
#pragma unroll
      for (int i = 0; i < 8; ++i) {
        int m = kh * 8 + i;
        int qrow = q0 + m;
        float e0 = (kv0 + cn <= qrow)      ? (float)s0[i] * kScale : kNegBig;
        float e1 = (kv0 + 16 + cn <= qrow) ? (float)s1[i] * kScale : kNegBig;
        float mx    = reduce_max16(fmaxf(e0, e1));
        float mnew  = fmaxf(mrow[i], mx);
        float alpha = __expf(mrow[i] - mnew);
        float p0 = __expf(e0 - mnew);
        float p1 = __expf(e1 - mnew);
        float ps = reduce_sum16(p0 + p1);
        lrow[i] = lrow[i] * alpha + ps;
        mrow[i] = mnew;
#pragma unroll
        for (int og = 0; og < 8; ++og) o[og][i] *= alpha;
        Ps[w][m][cn]      = (_Float16)p0;
        Ps[w][m][16 + cn] = (_Float16)p1;
      }
      // Wave-private LDS bounce: DS ops from one wave are in-order; fence
      // outstanding DS writes (and compiler reordering) before the gather.
      wait_ds0();

      // ---- P (16x32) as A fragment ----
      half16 pa;
#pragma unroll
      for (int j = 0; j < 8; ++j) {
        int kk = ((j >> 2) << 4) + kh * 8 + ((j & 3) << 1);
        pa[2 * j]     = Ps[w][cn][kk];
        pa[2 * j + 1] = Ps[w][cn][kk + 1];
      }

      // ---- O += P * V  (8 output column groups) ----
#pragma unroll
      for (int og = 0; og < 8; ++og) {
        half16 bv;
#pragma unroll
        for (int j = 0; j < 8; ++j) {
          int kk = ((j >> 2) << 4) + kh * 8 + ((j & 3) << 1);
          bv[2 * j]     = Vs[buf][kk][og * 16 + cn];
          bv[2 * j + 1] = Vs[buf][kk + 1][og * 16 + cn];
        }
        o[og] = WMMA_F16F32(pa, bv, o[og]);
      }
    }
  }

  // ---- epilogue: normalize and store context (f16) ----
  _Float16* cb = ctx + ((size_t)bb * S + q0) * kD + headOff;
#pragma unroll
  for (int og = 0; og < 8; ++og)
#pragma unroll
    for (int i = 0; i < 8; ++i) {
      int m = kh * 8 + i;
      float val = o[og][i] / lrow[i];
      cb[(size_t)m * kD + og * 16 + cn] = (_Float16)val;
    }
}

// ---------------------------------------------------------------------------
// Launch. Inputs (f32): x, wq, wk, wv, wo, attn_mask (mask unused: causal
// computed analytically). Output: f32 [B,S,D]. Workspace layout (f16):
//   xh(8M) wq(4M) wk(4M) wv(4M) wo(4M) q(8M) k(8M) v(8M) ctx(8M)  ~112 MB.
// ---------------------------------------------------------------------------
extern "C" void kernel_launch(void* const* d_in, const int* in_sizes, int n_in,
                              void* d_out, int out_size, void* d_ws, size_t ws_size,
                              hipStream_t stream) {
  (void)in_sizes; (void)n_in; (void)out_size; (void)ws_size;

  const float* x  = (const float*)d_in[0];
  const float* wq = (const float*)d_in[1];
  const float* wk = (const float*)d_in[2];
  const float* wv = (const float*)d_in[3];
  const float* wo = (const float*)d_in[4];

  const int BS = kB * kS;                 // 4096
  const size_t nX = (size_t)BS * kD;      // 8,388,608
  const size_t nW = (size_t)kD * kD;      // 4,194,304

  _Float16* ws  = (_Float16*)d_ws;
  _Float16* xh  = ws;
  _Float16* wqh = xh + nX;
  _Float16* wkh = wqh + nW;
  _Float16* wvh = wkh + nW;
  _Float16* woh = wvh + nW;
  _Float16* qh  = woh + nW;
  _Float16* kh  = qh + nX;
  _Float16* vh  = kh + nX;
  _Float16* ch  = vh + nX;

  // 1) casts to f16
  cast_f32_to_f16<<<2048, 256, 0, stream>>>(x,  xh,  (int)nX);
  cast_f32_to_f16<<<1024, 256, 0, stream>>>(wq, wqh, (int)nW);
  cast_f32_to_f16<<<1024, 256, 0, stream>>>(wk, wkh, (int)nW);
  cast_f32_to_f16<<<1024, 256, 0, stream>>>(wv, wvh, (int)nW);
  cast_f32_to_f16<<<1024, 256, 0, stream>>>(wo, woh, (int)nW);

  // 2) QKV projections (C = X * W^T)
  dim3 gGemm(kD / 128, BS / 128);
  gemm_nt_wmma<_Float16><<<gGemm, 128, 0, stream>>>(xh, wqh, qh, BS, kD, kD);
  gemm_nt_wmma<_Float16><<<gGemm, 128, 0, stream>>>(xh, wkh, kh, BS, kD, kD);
  gemm_nt_wmma<_Float16><<<gGemm, 128, 0, stream>>>(xh, wvh, vh, BS, kD, kD);

  // 3) RoPE on Q and K
  const int nPairs = kB * kS * kH * (kHD / 2);  // 4,194,304
  rope_kernel<<<2048, 256, 0, stream>>>(qh, kS, nPairs);
  rope_kernel<<<2048, 256, 0, stream>>>(kh, kS, nPairs);

  // 4) causal flash attention
  dim3 gAttn(kS / 64, kH, kB);
  attn_fwd<<<gAttn, 128, 0, stream>>>(qh, kh, vh, ch, kS);

  // 5) output projection, f32 out
  gemm_nt_wmma<float><<<gGemm, 128, 0, stream>>>(ch, woh, (float*)d_out, BS, kD, kD);
}
